// CrossAttention_52896817217654
// MI455X (gfx1250) — compile-verified, matrix operand order checked by reading, not run on hardware
//
#include <hip/hip_runtime.h>

// ---------------------------------------------------------------------------
// Cross-attention (B=8192, H=8, E=256, PH=DK=DV=32, SQ=5, SK=49), fp32 in/out.
// Memory-bound (AI ~23 flop/B -> ~41us floor at 23.3 TB/s). Strategy:
//   * 1 workgroup (256 thr = 8 wave32) per batch, 1 wave per head.
//   * Stage all per-batch inputs once into LDS as bf16 (zero-padded tiles).
//   * All GEMMs via v_wmma_f32_16x16x32_bf16 (bf16 in, f32 accum).
//   * Wo pre-converted to bf16^T in d_ws by a prep kernel (L2-resident).
//   * Invariant WMMA fragments hoisted into registers (no redundant ds loads).
// ---------------------------------------------------------------------------

typedef __bf16 bf16;
typedef __attribute__((ext_vector_type(16))) __bf16 bf16x16;
typedef __attribute__((ext_vector_type(8)))  __bf16 bf16x8;
typedef __attribute__((ext_vector_type(8)))  float  f32x8;

#define SQ_ 5
#define SK_ 49
#define H_  8
#define E_  256
#define PH_ 32

#define LD_IN 264   // row stride (bf16 elems) for staged [*,256] buffers (+8 pad)
#define LD_T  72    // row stride for per-head 64-wide tiles (+8 pad)

// A-style fragment load (16x32 bf16 tile). Per ISA 7.12.2 (16-bit A 16x32):
// lane l: m = l%16, half g = l/16; elem e -> K = (e/8)*16 + g*8 + (e%8).
// Two contiguous 16B runs per lane -> 2x ds_load_b128 / global_load_b128.
__device__ __forceinline__ bf16x16 frag_ld(const bf16* p, int ld, int row0, int col0) {
  const int l = threadIdx.x & 31;
  const int m = l & 15;
  const int g = l >> 4;
  const bf16* q = p + (row0 + m) * ld + col0 + g * 8;
  bf16x8 lo = *(const bf16x8*)(q);
  bf16x8 hi = *(const bf16x8*)(q + 16);
  bf16x16 f;
#pragma unroll
  for (int i = 0; i < 8; ++i) { f[i] = lo[i]; f[i + 8] = hi[i]; }
  return f;
}

// D (16x16 f32) store to bf16 row-major: lane l holds col n=l%16, rows g*8+r.
__device__ __forceinline__ void frag_st(bf16* p, int ld, int row0, int col0, f32x8 d) {
  const int l = threadIdx.x & 31;
  const int n = l & 15;
  const int g = l >> 4;
  bf16* q = p + (row0 + g * 8) * ld + col0 + n;
#pragma unroll
  for (int r = 0; r < 8; ++r) q[r * ld] = (bf16)d[r];
}

// D store transposed (dst[n][m] = D[m][n]) -> one contiguous 16B store/lane.
__device__ __forceinline__ void frag_st_T(bf16* p, int ld, int nrow0, int mcol0, f32x8 d) {
  const int l = threadIdx.x & 31;
  const int n = l & 15;
  const int g = l >> 4;
  bf16* q = p + (nrow0 + n) * ld + mcol0 + g * 8;
  bf16x8 v;
#pragma unroll
  for (int r = 0; r < 8; ++r) v[r] = (bf16)d[r];
  *(bf16x8*)q = v;
}

__device__ __forceinline__ f32x8 wmma_bf16(bf16x16 a, bf16x16 b, f32x8 c) {
  // 8 args: (neg_a, A, neg_b, B, c_mod, C, reuse_a, reuse_b)
  return __builtin_amdgcn_wmma_f32_16x16x32_bf16(false, a, false, b, (short)0, c,
                                                 false, false);
}

// One-time Wo [256,256] f32 -> WoT [n][k] bf16 (B-operand friendly layout).
__global__ void __launch_bounds__(256)
prep_wot_kernel(const float* __restrict__ Wo, bf16* __restrict__ WoT) {
  const int idx = blockIdx.x * 256 + threadIdx.x;   // 65536 threads total
  const int k = idx >> 8, n = idx & 255;
  WoT[n * 256 + k] = (bf16)Wo[k * 256 + n];
}

__global__ void __launch_bounds__(256)
cross_attn_kernel(const float* __restrict__ objq, const float* __restrict__ qpos,
                  const float* __restrict__ enc,  const float* __restrict__ kpos,
                  const float* __restrict__ Wq,   const float* __restrict__ Wk,
                  const float* __restrict__ Wv,   const bf16* __restrict__ WoT,
                  float* __restrict__ out) {
  extern __shared__ char smem_raw[];
  bf16* sQ   = (bf16*)smem_raw;            // 16 x 264  (rows 5..15 zero)
  bf16* sQP  = sQ   + 16 * LD_IN;          // 16 x 264
  bf16* sE   = sQP  + 16 * LD_IN;          // 64 x 264  (rows 49..63 zero)
  bf16* sKP  = sE   + 64 * LD_IN;          // 64 x 264
  bf16* sWqT = sKP  + 64 * LD_IN;          // 32 x 32 (transposed: [n][k])
  bf16* sWkT = sWqT + 1024;
  bf16* sWvT = sWkT + 1024;
  bf16* kbuf = sWvT + 1024;                // 8 heads x 64 x 72  (k_ row-major)
  bf16* vtb  = kbuf + 8 * 64 * LD_T;       // 8 heads x 32 x 72  (v transposed)
  bf16* qbuf = vtb  + 8 * 32 * LD_T;       // 8 heads x 16 x 72  (query)
  bf16* abuf = qbuf + 8 * 16 * LD_T;       // 8 heads x 16 x 72  (attn, bf16)
  bf16* sO   = abuf + 8 * 16 * LD_T;       // 16 x 264 (concat head outputs)
  float* sS  = (float*)(sO + 16 * LD_IN);  // 8 heads x 16 x 64 (scores, f32)

  const int tid  = threadIdx.x;
  const int lane = tid & 31;
  const int h    = tid >> 5;               // wave id == head id
  const int b    = blockIdx.x;
  const int hc   = h * PH_;

  const float* gq  = objq + (size_t)b * (SQ_ * E_);
  const float* gqp = qpos + (size_t)b * (SQ_ * E_);
  const float* ge  = enc  + (size_t)b * (SK_ * E_);
  const float* gkp = kpos + (size_t)b * (SK_ * E_);

  // ---- P0: zero staging region (pads -> exact zero tiles), then fill ----
  {
    uint32_t* z = (uint32_t*)smem_raw;     // sQ..sKP = 84480 B = 21120 dwords
    for (int i = tid; i < 21120; i += 256) z[i] = 0u;
  }
  __syncthreads();
  for (int i = tid; i < SQ_ * E_; i += 256) {
    const int m = i >> 8, e = i & 255;
    sQ [m * LD_IN + e] = (bf16)gq[i];
    sQP[m * LD_IN + e] = (bf16)gqp[i];
  }
  for (int i = tid; i < SK_ * E_; i += 256) {
    const int m = i >> 8, e = i & 255;
    sE [m * LD_IN + e] = (bf16)ge[i];
    sKP[m * LD_IN + e] = (bf16)gkp[i];
  }
  for (int i = tid; i < 1024; i += 256) {  // weights, stored transposed
    const int k = i >> 5, n = i & 31;
    sWqT[n * 32 + k] = (bf16)Wq[i];
    sWkT[n * 32 + k] = (bf16)Wk[i];
    sWvT[n * 32 + k] = (bf16)Wv[i];
  }
  __syncthreads();

  bf16*  kb = kbuf + h * 64 * LD_T;
  bf16*  vt = vtb  + h * 32 * LD_T;
  bf16*  qb = qbuf + h * 16 * LD_T;
  bf16*  ab = abuf + h * 16 * LD_T;
  float* sc = sS   + h * 16 * 64;

  // ---- P1 (per wave/head): query = [q@Wq | qp], k_ = [enc@Wk | kp], v^T ----
  {
    // Weight B-fragments: load once, reuse across all row tiles.
    const bf16x16 wq0 = frag_ld(sWqT, 32, 0, 0),  wq1 = frag_ld(sWqT, 32, 16, 0);
    const bf16x16 wk0 = frag_ld(sWkT, 32, 0, 0),  wk1 = frag_ld(sWkT, 32, 16, 0);
    const bf16x16 wv0 = frag_ld(sWvT, 32, 0, 0),  wv1 = frag_ld(sWvT, 32, 16, 0);

    const bf16x16 aq = frag_ld(sQ, LD_IN, 0, hc);
    f32x8 c0 = {}, c1 = {};
    c0 = wmma_bf16(aq, wq0, c0);
    c1 = wmma_bf16(aq, wq1, c1);
    frag_st(qb, LD_T, 0, 0,  c0);
    frag_st(qb, LD_T, 0, 16, c1);

#pragma unroll
    for (int mt = 0; mt < 4; ++mt) {
      const bf16x16 a = frag_ld(sE, LD_IN, mt * 16, hc);
      f32x8 k0 = {}, k1 = {}, v0 = {}, v1 = {};
      k0 = wmma_bf16(a, wk0, k0);
      k1 = wmma_bf16(a, wk1, k1);
      v0 = wmma_bf16(a, wv0, v0);
      v1 = wmma_bf16(a, wv1, v1);
      frag_st  (kb, LD_T, mt * 16, 0,  k0);       // k_ row-major [49(64),64]
      frag_st  (kb, LD_T, mt * 16, 16, k1);
      frag_st_T(vt, LD_T, 0,  mt * 16, v0);       // v^T [32,64]
      frag_st_T(vt, LD_T, 16, mt * 16, v1);
    }
  }
  // qp -> query cols 32..63 (dword copies; rows 5..15 are zeros from sQP pad)
  for (int i = lane; i < 16 * 16; i += 32) {
    const int m = i >> 4, j = (i & 15) * 2;
    *(uint32_t*)(qb + m * LD_T + 32 + j) =
        *(const uint32_t*)(sQP + m * LD_IN + hc + j);
  }
  // kp -> k_ cols 32..63 (rows 49..63 get zeros from sKP pad)
  for (int i = lane; i < 64 * 16; i += 32) {
    const int m = i >> 4, j = (i & 15) * 2;
    *(uint32_t*)(kb + m * LD_T + 32 + j) =
        *(const uint32_t*)(sKP + m * LD_IN + hc + j);
  }
  // clean attn tile (pad rows/cols must be exact zeros)
  for (int i = lane; i < (16 * LD_T) / 2; i += 32)
    ((uint32_t*)ab)[i] = 0u;
  __syncthreads();

  // ---- P2: scores[5,49] = query @ k_^T / sqrt(E) (k_ rows == B-operand) ----
  {
    const bf16x16 a0 = frag_ld(qb, LD_T, 0, 0);
    const bf16x16 a1 = frag_ld(qb, LD_T, 0, 32);
    const int n = lane & 15, g = lane >> 4;
#pragma unroll
    for (int t = 0; t < 4; ++t) {
      f32x8 acc = {};
      acc = wmma_bf16(a0, frag_ld(kb, LD_T, t * 16, 0),  acc);
      acc = wmma_bf16(a1, frag_ld(kb, LD_T, t * 16, 32), acc);
#pragma unroll
      for (int r = 0; r < 8; ++r)
        sc[(g * 8 + r) * 64 + t * 16 + n] = acc[r] * 0.0625f;  // 1/sqrt(256)
    }
  }
  __syncthreads();

  // ---- P3: softmax over k<49 (rows 0..4), write bf16 attn (cols>=49 stay 0) --
  if (lane < SQ_) {
    float* row = sc + lane * 64;
    float mx = -3.4e38f;
    for (int k = 0; k < SK_; ++k) mx = fmaxf(mx, row[k]);
    float sum = 0.f;
    for (int k = 0; k < SK_; ++k) {        // cache exp in the f32 score row
      const float e = __expf(row[k] - mx);
      row[k] = e;
      sum += e;
    }
    const float inv = 1.f / sum;
    bf16* arow = ab + lane * LD_T;
    for (int k = 0; k < SK_; ++k) arow[k] = (bf16)(row[k] * inv);
  }
  __syncthreads();

  // ---- P4: o = attn @ v  (v^T rows == B-operand) -> sO cols h*32.. ----
  {
    const bf16x16 a0 = frag_ld(ab, LD_T, 0, 0);
    const bf16x16 a1 = frag_ld(ab, LD_T, 0, 32);
#pragma unroll
    for (int nt = 0; nt < 2; ++nt) {
      f32x8 acc = {};
      acc = wmma_bf16(a0, frag_ld(vt, LD_T, nt * 16, 0),  acc);
      acc = wmma_bf16(a1, frag_ld(vt, LD_T, nt * 16, 32), acc);
      frag_st(sO, LD_IN, 0, hc + nt * 16, acc);
    }
  }
  __syncthreads();

  // ---- P5: out[5,256] = sO[5,256] @ Wo ; wave h owns N-tiles 2h, 2h+1 ----
  {
    bf16x16 aO[8];
#pragma unroll
    for (int ks = 0; ks < 8; ++ks) aO[ks] = frag_ld(sO, LD_IN, 0, ks * 32);

    float* gout = out + (size_t)b * (SQ_ * E_);
#pragma unroll
    for (int tt = 0; tt < 2; ++tt) {
      const int n0 = (2 * h + tt) * 16;
      f32x8 acc = {};
#pragma unroll
      for (int ks = 0; ks < 8; ++ks)
        acc = wmma_bf16(aO[ks], frag_ld(WoT, 256, n0, ks * 32), acc);
      if (lane < 16) {                       // rows m = r (g==0), only m<5 valid
#pragma unroll
        for (int r = 0; r < SQ_; ++r)
          gout[r * E_ + n0 + lane] = acc[r];
      }
    }
  }
}

extern "C" void kernel_launch(void* const* d_in, const int* in_sizes, int n_in,
                              void* d_out, int out_size, void* d_ws, size_t ws_size,
                              hipStream_t stream) {
  const float* objq = (const float*)d_in[0];
  const float* qpos = (const float*)d_in[1];
  const float* enc  = (const float*)d_in[2];
  const float* kpos = (const float*)d_in[3];
  const float* Wq   = (const float*)d_in[4];
  const float* Wk   = (const float*)d_in[5];
  const float* Wv   = (const float*)d_in[6];
  const float* Wo   = (const float*)d_in[7];
  float* out = (float*)d_out;
  bf16* WoT  = (bf16*)d_ws;                 // needs 256*256*2 = 128 KB scratch

  const int B = in_sizes[0] / (SQ_ * E_);

  prep_wot_kernel<<<256, 256, 0, stream>>>(Wo, WoT);

  // Dynamic LDS: staged inputs + per-head tiles + scores = 279,296 B (<320KB/WGP)
  const int smem_bytes = 279296;
  cross_attn_kernel<<<B, 256, smem_bytes, stream>>>(objq, qpos, enc, kpos,
                                                    Wq, Wk, Wv, WoT, out);
}
